// RnntJoiner_4844723109998
// MI455X (gfx1250) — matrix operand order repaired and checked by reading, not compiled, not run on hardware
//
#include <hip/hip_runtime.h>
#include <hip/hip_bf16.h>

typedef _Float16 h8   __attribute__((ext_vector_type(8)));
typedef _Float16 v16h __attribute__((ext_vector_type(16)));
typedef float    v8f  __attribute__((ext_vector_type(8)));
typedef float    f4   __attribute__((ext_vector_type(4)));

// Problem dims (from reference setup_inputs)
constexpr int Bn = 8, Tn = 256, Un = 64, Dn = 512, Vn = 1024;
constexpr int TM  = 128;   // output rows per block: 2 t-values x 64 u
constexpr int TN  = 128;   // V columns per block
constexpr int KC  = 32;    // K chunk (f16 WMMA K)
constexpr int NK  = Dn / KC;
constexpr int LDA = 40;    // padded LDS row stride in halves (80B)
constexpr int LDB = 40;

union V16U { v16h v; h8 h[2]; };

static __device__ __forceinline__ h8 relu_cvt8(const f4 xa, const f4 xb,
                                               const f4 ya, const f4 yb) {
    h8 r;
    #pragma unroll
    for (int i = 0; i < 4; ++i) {
        float v0 = xa[i] + ya[i];
        float v1 = xb[i] + yb[i];
        v0 = v0 > 0.f ? v0 : 0.f;
        v1 = v1 > 0.f ? v1 : 0.f;
        r[i]     = (_Float16)v0;
        r[i + 4] = (_Float16)v1;
    }
    return r;
}

// CDNA5 async global->LDS copy (16B), GVS addressing, tracked by ASYNCcnt.
static __device__ __forceinline__ void async_b128(uint32_t ldsoff, uint64_t sbase,
                                                  uint32_t voff) {
    asm volatile("global_load_async_to_lds_b128 %0, %1, %2"
                 :: "v"(ldsoff), "v"(voff), "s"(sbase) : "memory");
}
static __device__ __forceinline__ void wait_async0() {
    asm volatile("s_wait_asynccnt 0x0" ::: "memory");
}

// One-time W f32 -> f16 pre-pass (W stays resident in the 192MB L2 as 1MB f16).
__global__ __launch_bounds__(256) void convert_w_f16(const float* __restrict__ w,
                                                     _Float16* __restrict__ wh) {
    const int i = (blockIdx.x * 256 + threadIdx.x) * 8;
    f4 a = *(const f4*)(w + i);
    f4 b = *(const f4*)(w + i + 4);
    h8 r;
    #pragma unroll
    for (int j = 0; j < 4; ++j) { r[j] = (_Float16)a[j]; r[j + 4] = (_Float16)b[j]; }
    *(h8*)(wh + i) = r;
}

template <bool PRE>
__global__ __launch_bounds__(256)
void rnnt_joiner_wmma(const float* __restrict__ x,    // [B,T,D]
                      const float* __restrict__ y,    // [B,U,D]
                      const float* __restrict__ Wf,   // [V,D] f32 (fallback)
                      const _Float16* __restrict__ Wh,// [V,D] f16 (pre-pass)
                      const float* __restrict__ bias, // [V]
                      float* __restrict__ out)        // [B,T,U,V]
{
    __shared__ __align__(16) _Float16 As[2][TM * LDA];  // relu(x+y) tiles
    __shared__ __align__(16) _Float16 Bs[2][TN * LDB];  // W tiles

    const int tid  = threadIdx.x;
    const int wave = tid >> 5;
    const int lane = tid & 31;

    const int n0  = blockIdx.x * TN;
    const int bty = blockIdx.y;                 // 0..1023
    const int bb  = bty >> 7;                   // / (T/2)
    const int t0  = (bty & 127) * 2;

    float* obase = out + ((size_t)(bb * Tn + t0) * Un) * Vn;  // rows m=0..127 contiguous

    // Wave tiling: 2 (M) x 4 (N); each wave owns 64x32 = 4x2 WMMA tiles.
    const int wm = wave >> 2;                   // 0..1  -> m base wm*64
    const int wn = wave & 3;                    // 0..3  -> n base wn*32

    v8f acc[4][2];
    #pragma unroll
    for (int mt = 0; mt < 4; ++mt)
        #pragma unroll
        for (int nt = 0; nt < 2; ++nt)
            acc[mt][nt] = (v8f){0.f,0.f,0.f,0.f,0.f,0.f,0.f,0.f};

    // Cooperative staging indices (256 threads, 16 elems each per tile)
    const int s_m = tid >> 1;                   // 0..127 (A row, also B row n)
    const int s_k = (tid & 1) * 16;             // 0 or 16
    const int a_u = s_m & 63;
    const int a_t = s_m >> 6;
    const float* ya = y + ((size_t)bb * Un + a_u) * Dn;
    const float* xa = x + ((size_t)(bb * Tn + t0 + a_t)) * Dn;
    const int   wrow = n0 + s_m;                // W row for B staging
    _Float16* asdst0 = &As[0][s_m * LDA + s_k];
    _Float16* asdst1 = &As[1][s_m * LDA + s_k];
    _Float16* bsdst0 = &Bs[0][s_m * LDB + s_k];
    _Float16* bsdst1 = &Bs[1][s_m * LDB + s_k];

    // Fragment indices (contiguous k: lane half h holds k = 16h..16h+15)
    const int frow = lane & 15;
    const int kbA  = (lane >> 4) * 16;
    const int kbB  = (lane >> 4) * 16;

    // ---------------- prologue: stage chunk 0 into buffer 0 ----------------
    {
        if constexpr (PRE) {
            const uint32_t l0 = (uint32_t)(uintptr_t)bsdst0;
            const uint32_t g0 = (uint32_t)(((size_t)wrow * Dn + s_k) * 2);
            async_b128(l0,      (uint64_t)(uintptr_t)Wh, g0);
            async_b128(l0 + 16, (uint64_t)(uintptr_t)Wh, g0 + 16);
        } else {
            const float* wp = Wf + (size_t)wrow * Dn + s_k;
            f4 w0 = *(const f4*)(wp);     f4 w1 = *(const f4*)(wp + 4);
            f4 w2 = *(const f4*)(wp + 8); f4 w3 = *(const f4*)(wp + 12);
            h8 bh0, bh1;
            #pragma unroll
            for (int i = 0; i < 4; ++i) {
                bh0[i] = (_Float16)w0[i]; bh0[i+4] = (_Float16)w1[i];
                bh1[i] = (_Float16)w2[i]; bh1[i+4] = (_Float16)w3[i];
            }
            *(h8*)(bsdst0)     = bh0;
            *(h8*)(bsdst0 + 8) = bh1;
        }
        f4 Y0 = *(const f4*)(ya + s_k);     f4 Y1 = *(const f4*)(ya + s_k + 4);
        f4 Y2 = *(const f4*)(ya + s_k + 8); f4 Y3 = *(const f4*)(ya + s_k + 12);
        f4 X0 = *(const f4*)(xa + s_k);     f4 X1 = *(const f4*)(xa + s_k + 4);
        f4 X2 = *(const f4*)(xa + s_k + 8); f4 X3 = *(const f4*)(xa + s_k + 12);
        *(h8*)(asdst0)     = relu_cvt8(X0, X1, Y0, Y1);
        *(h8*)(asdst0 + 8) = relu_cvt8(X2, X3, Y2, Y3);
        if constexpr (PRE) wait_async0();
    }
    __syncthreads();

    // ---------------- main loop: double-buffered, 1 barrier / chunk ----------------
    for (int kt = 0; kt < NK; ++kt) {
        const int  cur   = kt & 1;
        const bool more  = (kt + 1) < NK;
        const int  knext = (kt + 1) * KC;
        _Float16* const asdst = cur ? asdst0 : asdst1;   // next buffer
        _Float16* const bsdst = cur ? bsdst0 : bsdst1;

        f4 X0{}, X1{}, X2{}, X3{}, Y0{}, Y1{}, Y2{}, Y3{};
        f4 w0{}, w1{}, w2{}, w3{};
        if (more) {
            // issue next-chunk global traffic before the matrix work
            Y0 = *(const f4*)(ya + knext + s_k);     Y1 = *(const f4*)(ya + knext + s_k + 4);
            Y2 = *(const f4*)(ya + knext + s_k + 8); Y3 = *(const f4*)(ya + knext + s_k + 12);
            X0 = *(const f4*)(xa + knext + s_k);     X1 = *(const f4*)(xa + knext + s_k + 4);
            X2 = *(const f4*)(xa + knext + s_k + 8); X3 = *(const f4*)(xa + knext + s_k + 12);
            if constexpr (PRE) {
                const uint32_t l0 = (uint32_t)(uintptr_t)bsdst;
                const uint32_t g0 = (uint32_t)(((size_t)wrow * Dn + knext + s_k) * 2);
                async_b128(l0,      (uint64_t)(uintptr_t)Wh, g0);
                async_b128(l0 + 16, (uint64_t)(uintptr_t)Wh, g0 + 16);
            } else {
                const float* wp = Wf + (size_t)wrow * Dn + knext + s_k;
                w0 = *(const f4*)(wp);     w1 = *(const f4*)(wp + 4);
                w2 = *(const f4*)(wp + 8); w3 = *(const f4*)(wp + 12);
            }
        }

        // ---- fragments from current buffer ----
        const _Float16* asrc = As[cur];
        const _Float16* bsrc = Bs[cur];
        v16h afrag[4], bfrag[2];
        #pragma unroll
        for (int mt = 0; mt < 4; ++mt) {
            const int m = wm * 64 + mt * 16 + frow;
            V16U t;
            t.h[0] = *(const h8*)&asrc[m * LDA + kbA];        // K = kb..kb+7
            t.h[1] = *(const h8*)&asrc[m * LDA + kbA + 8];    // K = kb+8..kb+15
            afrag[mt] = t.v;
        }
        #pragma unroll
        for (int nt = 0; nt < 2; ++nt) {
            const int n = wn * 32 + nt * 16 + frow;
            V16U t;
            t.h[0] = *(const h8*)&bsrc[n * LDB + kbB];        // K = kb..kb+7
            t.h[1] = *(const h8*)&bsrc[n * LDB + kbB + 8];    // K = kb+8..kb+15
            bfrag[nt] = t.v;
        }

        // ---- 4x2 WMMA: D = A x B + C, f32 accumulate ----
        #pragma unroll
        for (int mt = 0; mt < 4; ++mt)
            #pragma unroll
            for (int nt = 0; nt < 2; ++nt)
                acc[mt][nt] = __builtin_amdgcn_wmma_f32_16x16x32_f16(
                    false, afrag[mt], false, bfrag[nt],
                    (short)0, acc[mt][nt], false, false);

        if (more) {
            // finish staging the next buffer
            *(h8*)(asdst)     = relu_cvt8(X0, X1, Y0, Y1);
            *(h8*)(asdst + 8) = relu_cvt8(X2, X3, Y2, Y3);
            if constexpr (!PRE) {
                h8 bh0, bh1;
                #pragma unroll
                for (int i = 0; i < 4; ++i) {
                    bh0[i] = (_Float16)w0[i]; bh0[i+4] = (_Float16)w1[i];
                    bh1[i] = (_Float16)w2[i]; bh1[i+4] = (_Float16)w3[i];
                }
                *(h8*)(bsdst)     = bh0;
                *(h8*)(bsdst + 8) = bh1;
            } else {
                wait_async0();
            }
            __syncthreads();
        }
    }

    // ---- epilogue: add bias, store (row = (lane>>4)*8 + r, col = lane&15) ----
    const int col   = lane & 15;
    const int rbase = (lane >> 4) * 8;
    #pragma unroll
    for (int nt = 0; nt < 2; ++nt) {
        const int v = n0 + wn * 32 + nt * 16 + col;
        const float bv = bias[v];
        #pragma unroll
        for (int mt = 0; mt < 4; ++mt) {
            const int mb = wm * 64 + mt * 16 + rbase;
            #pragma unroll
            for (int r = 0; r < 8; ++r) {
                obase[(size_t)(mb + r) * Vn + v] = acc[mt][nt][r] + bv;
            }
        }
    }
}

extern "C" void kernel_launch(void* const* d_in, const int* in_sizes, int n_in,
                              void* d_out, int out_size, void* d_ws, size_t ws_size,
                              hipStream_t stream) {
    (void)in_sizes; (void)n_in; (void)out_size;
    const float* x    = (const float*)d_in[0];
    const float* y    = (const float*)d_in[1];
    const float* Wf   = (const float*)d_in[2];
    const float* bias = (const float*)d_in[3];
    float*       out  = (float*)d_out;

    dim3 grid(Vn / TN, (Bn * Tn) / 2);   // (8, 1024)
    dim3 block(256);                     // 8 waves (wave32)

    const size_t whBytes = sizeof(_Float16) * (size_t)Vn * Dn;  // 1 MB
    if (ws_size >= whBytes) {
        _Float16* Wh = (_Float16*)d_ws;
        convert_w_f16<<<(Vn * Dn / 8) / 256, 256, 0, stream>>>(Wf, Wh);
        rnnt_joiner_wmma<true><<<grid, block, 0, stream>>>(x, y, Wf, Wh, bias, out);
    } else {
        rnnt_joiner_wmma<false><<<grid, block, 0, stream>>>(x, y, Wf, nullptr, bias, out);
    }
}
